// GNN_2551210574350
// MI455X (gfx1250) — matrix operand order repaired and checked by reading, not compiled, not run on hardware
//
#include <hip/hip_runtime.h>

// ---------------------------------------------------------------------------
// GNN forward (3x GraphConv + mean-pool + linear) for MI455X / gfx1250.
//   - GEMMs: v_wmma_f32_16x16x32_bf16, fp32 accumulate.
//   - Block = 64-row M-macro-tile x full N=256; features staged once in LDS
//     (conflict-free ds_load_b128 fragments); weight fragments loaded once
//     per k-step and reused across 4 M-subtiles (8 indep. WMMA chains).
//   - Scatter-add: wave/edge, vector bf16 gathers + global f32 atomics
//     (aggregate arrays are L2-resident: <= 51 MB << 192 MB L2).
// ---------------------------------------------------------------------------

typedef __bf16 bf16_t;
typedef __attribute__((ext_vector_type(16))) __bf16 v16bf;
typedef __attribute__((ext_vector_type(8)))  float  v8f;

#define N_NODES   50000
#define N_EDGES   800000
#define IN_CH     128
#define HID       256
#define N_CLASSES 10
#define N_GRAPHS  64

#define M_SUB     4            // 16-row subtiles per block (64-row macro-tile)
#define N_MTILES  ((N_NODES + 15) / 16)          // 3125

// ---------------- helpers: fragment assembly --------------------------------

__device__ __forceinline__ v16bf make_frag(int4 lo, int4 hi) {
  union { int4 q[2]; v16bf v; } u;
  u.q[0] = lo; u.q[1] = hi;
  return u.v;
}

// A-matrix 16x32 bf16 fragment (row-major source, row for this lane at col k0).
// Lanes 0-15: K in {half*8..+7, 16+half*8..+7}; works for LDS or global rows.
__device__ __forceinline__ v16bf load_frag_a(const bf16_t* row, int half) {
  int4 lo = *(const int4*)(row + half * 8);
  int4 hi = *(const int4*)(row + 16 + half * 8);
  return make_frag(lo, hi);
}

// B-matrix 32x16 bf16 fragment from pre-transposed weights WT[N][K];
// `row` = WT row for this lane's N at col k0; K in {half*8..+7, 16+half*8..+7}.
__device__ __forceinline__ v16bf load_frag_b(const bf16_t* __restrict__ row, int half) {
  const bf16_t* p = row + half * 8;
  int4 lo = *(const int4*)(p);
  int4 hi = *(const int4*)(p + 16);
  return make_frag(lo, hi);
}

#define WMMA_BF16(Afrag, Bfrag, Cacc) \
  __builtin_amdgcn_wmma_f32_16x16x32_bf16(false, (Afrag), false, (Bfrag), (short)0, (Cacc), false, false)

// ---------------- small utility kernels --------------------------------------

__global__ void cast_f32_bf16(const float* __restrict__ in, bf16_t* __restrict__ out, int n) {
  int i = blockIdx.x * blockDim.x + threadIdx.x;
  if (i < n) out[i] = (bf16_t)in[i];
}

// wt[n*K + k] = (bf16) w[k*N + n]  (transpose so B fragments are contiguous in K)
__global__ void transpose_cast(const float* __restrict__ w, bf16_t* __restrict__ wt,
                               int K, int N) {
  int idx = blockIdx.x * blockDim.x + threadIdx.x;
  if (idx >= K * N) return;
  int k = idx / N;
  int n = idx % N;
  wt[(size_t)n * K + k] = (bf16_t)w[idx];
}

// ---------------- edge scatter-add -------------------------------------------
// One wave32 per edge; each lane owns D/32 contiguous channels (vector gather).
template <int D>
__global__ void edge_agg_kernel(const bf16_t* __restrict__ feat,
                                const int* __restrict__ ei,  // [2, E] flat
                                float* __restrict__ agg, int E) {
  constexpr int PER = D / 32;                  // 4 (D=128) or 8 (D=256)
  const int lane = threadIdx.x & 31;
  const int e = blockIdx.x * (blockDim.x >> 5) + (threadIdx.x >> 5);
  if (e >= E) return;
  const int src = ei[e];
  const int dst = ei[E + e];
  const bf16_t* f = feat + (size_t)src * D + lane * PER;
  float* a = agg + (size_t)dst * D + lane * PER;

  union { int4 q; bf16_t h[8]; } u;
  if constexpr (PER == 8) {
    u.q = *(const int4*)f;                     // global_load_b128
  } else {
    int2 t = *(const int2*)f;                  // global_load_b64
    u.q.x = t.x; u.q.y = t.y;
  }
#pragma unroll
  for (int i = 0; i < PER; ++i) {
    atomicAdd(a + i, (float)u.h[i]);           // global_atomic_add_f32 (L2)
  }
}

// ---------------- fused dual-GEMM layer --------------------------------------
// OUT[M,256] = act( X[M,K] @ Wr[K,256] + A[M,K] @ Wn[K,256] + bias )
// Block = 8 waves; macro-tile = 64 rows x N=256; wave w owns N cols [32w,32w+32).
// Per k-step: 4 B-frags loaded once, reused across 4 M-subtiles -> 16 WMMAs.
template <int K, bool RELU>
__global__ void __launch_bounds__(256)
gemm_dual_wmma(const bf16_t* __restrict__ X, const bf16_t* __restrict__ A,
               const bf16_t* __restrict__ WrT, const bf16_t* __restrict__ WnT,
               const float* __restrict__ bias,
               bf16_t* __restrict__ out_b, float* __restrict__ out_f) {
  constexpr int N    = HID;                    // 256
  constexpr int LDK  = K + 8;                  // +16B pad -> bank-conflict-free
  constexpr int ROWS = 16 * M_SUB;             // 64
  __shared__ __align__(16) bf16_t sX[ROWS * LDK];
  __shared__ __align__(16) bf16_t sA[ROWS * LDK];

  const int tid  = threadIdx.x;
  const int wave = tid >> 5;
  const int lane = tid & 31;
  const int half = lane >> 4;
  const int l16  = lane & 15;
  const int m0   = blockIdx.x * ROWS;

  // Cooperative stage of the 64-row feature strip (X and AGG) into LDS.
  // Out-of-range rows (only in the final block) are zero-filled: zeros are
  // WMMA-safe and the stores below are guarded.
  constexpr int RCH = K / 8;                   // int4 chunks per row
#pragma unroll
  for (int idx = tid; idx < ROWS * RCH; idx += 256) {
    const int row = idx / RCH;
    const int col = (idx % RCH) * 8;
    int4 vx = {0, 0, 0, 0}, va = {0, 0, 0, 0};
    if (m0 + row < N_NODES) {
      vx = *(const int4*)(X + (size_t)(m0 + row) * K + col);
      va = *(const int4*)(A + (size_t)(m0 + row) * K + col);
    }
    *(int4*)(sX + row * LDK + col) = vx;
    *(int4*)(sA + row * LDK + col) = va;
  }
  __syncthreads();

  const int n0 = wave * 32;                    // this wave's 32-column slice
  const bf16_t* wr0 = WrT + (size_t)(n0 + l16) * K;
  const bf16_t* wr1 = WrT + (size_t)(n0 + 16 + l16) * K;
  const bf16_t* wn0 = WnT + (size_t)(n0 + l16) * K;
  const bf16_t* wn1 = WnT + (size_t)(n0 + 16 + l16) * K;

  __builtin_prefetch(wr0, 0, 3);               // global_prefetch_b8
  __builtin_prefetch(wr1, 0, 3);
  __builtin_prefetch(wn0, 0, 3);
  __builtin_prefetch(wn1, 0, 3);

  v8f acc[M_SUB][2] = {};
#pragma unroll 1
  for (int k0 = 0; k0 < K; k0 += 32) {
    const v16bf b0 = load_frag_b(wr0 + k0, half);   // 2x global_load_b128 each,
    const v16bf b1 = load_frag_b(wr1 + k0, half);   // loaded ONCE per k-step
    const v16bf b2 = load_frag_b(wn0 + k0, half);
    const v16bf b3 = load_frag_b(wn1 + k0, half);
#pragma unroll
    for (int s = 0; s < M_SUB; ++s) {
      const bf16_t* sxr = sX + (s * 16 + l16) * LDK + k0;
      const bf16_t* sar = sA + (s * 16 + l16) * LDK + k0;
      const v16bf ax = load_frag_a(sxr, half);      // 2x ds_load_b128
      const v16bf aa = load_frag_a(sar, half);
      acc[s][0] = WMMA_BF16(ax, b0, acc[s][0]);     // 8 independent chains
      acc[s][1] = WMMA_BF16(ax, b1, acc[s][1]);
      acc[s][0] = WMMA_BF16(aa, b2, acc[s][0]);
      acc[s][1] = WMMA_BF16(aa, b3, acc[s][1]);
    }
  }

  const float bv0 = bias[n0 + l16];
  const float bv1 = bias[n0 + 16 + l16];
#pragma unroll
  for (int s = 0; s < M_SUB; ++s) {
#pragma unroll
    for (int r = 0; r < 8; ++r) {
      const int m = m0 + s * 16 + half * 8 + r;     // C/D layout: M = 8*half + r
      if (m >= N_NODES) continue;
      float v0 = acc[s][0][r] + bv0;
      float v1 = acc[s][1][r] + bv1;
      if (RELU) {
        v0 = v0 > 0.0f ? v0 : 0.0f;
        v1 = v1 > 0.0f ? v1 : 0.0f;
      }
      if (out_b) {
        out_b[(size_t)m * N + n0 + l16]      = (bf16_t)v0;
        out_b[(size_t)m * N + n0 + 16 + l16] = (bf16_t)v1;
      }
      if (out_f) {
        out_f[(size_t)m * N + n0 + l16]      = v0;
        out_f[(size_t)m * N + n0 + 16 + l16] = v1;
      }
    }
  }
}

// ---------------- mean-pool (atomic sums + counts) ---------------------------

__global__ void pool_sum_kernel(const float* __restrict__ h, const int* __restrict__ batch,
                                float* __restrict__ sums, float* __restrict__ counts) {
  int idx = blockIdx.x * blockDim.x + threadIdx.x;     // N_NODES * 64 threads
  int node = idx >> 6;
  int part = idx & 63;                                 // 4 channels each
  if (node >= N_NODES) return;
  int g = batch[node];
  const float4 v = *(const float4*)(h + (size_t)node * HID + part * 4);
  float* s = sums + (size_t)g * HID + part * 4;
  atomicAdd(s + 0, v.x);
  atomicAdd(s + 1, v.y);
  atomicAdd(s + 2, v.z);
  atomicAdd(s + 3, v.w);
  if (part == 0) atomicAdd(&counts[g], 1.0f);
}

__global__ void final_linear(const float* __restrict__ sums, const float* __restrict__ counts,
                             const float* __restrict__ Wlin, const float* __restrict__ blin,
                             float* __restrict__ out) {
  int idx = blockIdx.x * blockDim.x + threadIdx.x;
  if (idx >= N_GRAPHS * N_CLASSES) return;
  int g = idx / N_CLASSES;
  int c = idx % N_CLASSES;
  float cnt = counts[g];
  cnt = cnt > 1.0f ? cnt : 1.0f;
  const float inv = 1.0f / cnt;
  float s = 0.0f;
#pragma unroll 4
  for (int k = 0; k < HID; ++k)
    s += sums[(size_t)g * HID + k] * inv * Wlin[k * N_CLASSES + c];
  out[idx] = s + blin[c];
}

// ---------------- host orchestration -----------------------------------------

extern "C" void kernel_launch(void* const* d_in, const int* in_sizes, int n_in,
                              void* d_out, int out_size, void* d_ws, size_t ws_size,
                              hipStream_t stream) {
  const float* x     = (const float*)d_in[0];
  const int*   ei    = (const int*)d_in[1];
  const int*   batch = (const int*)d_in[2];
  const float* W1r = (const float*)d_in[3];
  const float* W1n = (const float*)d_in[4];
  const float* b1  = (const float*)d_in[5];
  const float* W2r = (const float*)d_in[6];
  const float* W2n = (const float*)d_in[7];
  const float* b2  = (const float*)d_in[8];
  const float* W3r = (const float*)d_in[9];
  const float* W3n = (const float*)d_in[10];
  const float* b3  = (const float*)d_in[11];
  const float* Wl  = (const float*)d_in[12];
  const float* bl  = (const float*)d_in[13];

  char* ws = (char*)d_ws;
  size_t off = 0;
  auto take = [&](size_t bytes) -> char* {
    char* p = ws + off;
    off += (bytes + 255) & ~(size_t)255;
    return p;
  };
  bf16_t* featA = (bf16_t*)take((size_t)N_NODES * HID * 2);
  bf16_t* featB = (bf16_t*)take((size_t)N_NODES * HID * 2);
  bf16_t* aggb  = (bf16_t*)take((size_t)N_NODES * HID * 2);
  float*  aggf  = (float*) take((size_t)N_NODES * HID * 4);   // reused as layer-3 fp32 out
  float*  sums  = (float*) take((size_t)N_GRAPHS * HID * 4);
  float*  cnts  = (float*) take((size_t)N_GRAPHS * 4);
  bf16_t* W1rT = (bf16_t*)take((size_t)IN_CH * HID * 2);
  bf16_t* W1nT = (bf16_t*)take((size_t)IN_CH * HID * 2);
  bf16_t* W2rT = (bf16_t*)take((size_t)HID * HID * 2);
  bf16_t* W2nT = (bf16_t*)take((size_t)HID * HID * 2);
  bf16_t* W3rT = (bf16_t*)take((size_t)HID * HID * 2);
  bf16_t* W3nT = (bf16_t*)take((size_t)HID * HID * 2);

  const int T = 256;
  const int gw1 = (IN_CH * HID + T - 1) / T;
  const int gw2 = (HID * HID + T - 1) / T;
  transpose_cast<<<gw1, T, 0, stream>>>(W1r, W1rT, IN_CH, HID);
  transpose_cast<<<gw1, T, 0, stream>>>(W1n, W1nT, IN_CH, HID);
  transpose_cast<<<gw2, T, 0, stream>>>(W2r, W2rT, HID, HID);
  transpose_cast<<<gw2, T, 0, stream>>>(W2n, W2nT, HID, HID);
  transpose_cast<<<gw2, T, 0, stream>>>(W3r, W3rT, HID, HID);
  transpose_cast<<<gw2, T, 0, stream>>>(W3n, W3nT, HID, HID);

  const int n_in1 = N_NODES * IN_CH;
  const int n_hid = N_NODES * HID;
  cast_f32_bf16<<<(n_in1 + T - 1) / T, T, 0, stream>>>(x, featA, n_in1);

  const int edge_blocks = (N_EDGES + 7) / 8;                  // 8 waves/block
  const int gemm_blocks = (N_MTILES + M_SUB - 1) / M_SUB;     // 782 macro-tiles

  // ---- layer 1: K = 128, ReLU, bf16 out -> featB
  hipMemsetAsync(aggf, 0, (size_t)n_in1 * 4, stream);
  edge_agg_kernel<IN_CH><<<edge_blocks, T, 0, stream>>>(featA, ei, aggf, N_EDGES);
  cast_f32_bf16<<<(n_in1 + T - 1) / T, T, 0, stream>>>(aggf, aggb, n_in1);
  gemm_dual_wmma<IN_CH, true><<<gemm_blocks, T, 0, stream>>>(
      featA, aggb, W1rT, W1nT, b1, featB, nullptr);

  // ---- layer 2: K = 256, ReLU, bf16 out -> featA
  hipMemsetAsync(aggf, 0, (size_t)n_hid * 4, stream);
  edge_agg_kernel<HID><<<edge_blocks, T, 0, stream>>>(featB, ei, aggf, N_EDGES);
  cast_f32_bf16<<<(n_hid + T - 1) / T, T, 0, stream>>>(aggf, aggb, n_hid);
  gemm_dual_wmma<HID, true><<<gemm_blocks, T, 0, stream>>>(
      featB, aggb, W2rT, W2nT, b2, featA, nullptr);

  // ---- layer 3: K = 256, no ReLU, fp32 out -> aggf (GEMM reads aggb, safe)
  hipMemsetAsync(aggf, 0, (size_t)n_hid * 4, stream);
  edge_agg_kernel<HID><<<edge_blocks, T, 0, stream>>>(featA, ei, aggf, N_EDGES);
  cast_f32_bf16<<<(n_hid + T - 1) / T, T, 0, stream>>>(aggf, aggb, n_hid);
  float* h3 = aggf;
  gemm_dual_wmma<HID, false><<<gemm_blocks, T, 0, stream>>>(
      featA, aggb, W3rT, W3nT, b3, nullptr, h3);

  // ---- global mean pool + classifier head
  hipMemsetAsync(sums, 0, (size_t)N_GRAPHS * HID * 4, stream);
  hipMemsetAsync(cnts, 0, (size_t)N_GRAPHS * 4, stream);
  pool_sum_kernel<<<(N_NODES * 64 + T - 1) / T, T, 0, stream>>>(h3, batch, sums, cnts);
  final_linear<<<(N_GRAPHS * N_CLASSES + T - 1) / T, T, 0, stream>>>(
      sums, cnts, Wl, bl, (float*)d_out);
}
